// Attention_24309514896115
// MI455X (gfx1250) — matrix operand order repaired and checked by reading, not compiled, not run on hardware
//
#include <hip/hip_runtime.h>
#include <hip/hip_bf16.h>
#include <math.h>

// ---------------------------------------------------------------------------
// Bahdanau attention for MI455X (gfx1250, wave32, WMMA).
//   S=2048, B=16, H=1024
//   energy = tanh( enc @ W2^T  +  (hid @ W1^T + bias) )   [fused, never stored]
//   scores = energy @ v ; attn = softmax_s ; ctx = attn @ enc
// Dominant GEMM (32768x1024)x(1024x1024) on v_wmma_f32_16x16x32_f16, f32 acc.
// A staged in 256KB LDS in kt-block-major order so all 8 M-tile fragment loads
// use DS immediate offsets from one base register (minimal address VALU).
// B pre-swizzled into per-lane WMMA fragment order in workspace.
// ---------------------------------------------------------------------------

typedef __attribute__((ext_vector_type(16))) _Float16 v16h;
typedef __attribute__((ext_vector_type(8)))  float    v8f;

#define S_DIM 2048
#define B_DIM 16
#define H_DIM 1024
#define MBLK  128          // rows of (s,b) per workgroup
#define KTILES 32          // 1024 / 32 (K per wmma)
#define NTILES 64          // 1024 / 16 (output cols)

union Frag16 { uint4 q[2]; v16h h; };

// ---------------------------------------------------------------------------
// Kernel 0: hb[b,h] = bias[h] + dot(hidden[b,:], W[h, 0:H])    (16384 threads)
// ---------------------------------------------------------------------------
__global__ __launch_bounds__(256) void hbias_kernel(
    const float* __restrict__ hidden, const float* __restrict__ W,
    const float* __restrict__ bias, float* __restrict__ hb) {
  const int t = blockIdx.x * 256 + threadIdx.x;          // 0..16383
  const int b = t >> 10, h = t & (H_DIM - 1);
  const float* hp = hidden + (size_t)b * H_DIM;
  const float* wp = W + (size_t)h * (2 * H_DIM);         // W row h, first half
  float acc = bias[h];
#pragma unroll 4
  for (int k = 0; k < H_DIM; k += 4) {
    const float4 x = *(const float4*)(hp + k);
    const float4 w = *(const float4*)(wp + k);
    acc += x.x * w.x + x.y * w.y + x.z * w.z + x.w * w.w;
  }
  hb[t] = acc;
}

// ---------------------------------------------------------------------------
// Kernel 1: pack W2 = W[:, H:2H] into f16 B-fragments.
// k-contiguous B-layout used here (16-bit, 32x16): lane L -> col n=L&15, half=L>>4;
// dword g (0..7) holds K = half*16 + 2g (+1). One dword per thread.
// ---------------------------------------------------------------------------
__global__ __launch_bounds__(256) void packB_kernel(
    const float* __restrict__ W, _Float16* __restrict__ Bpack) {
  const int idx  = blockIdx.x * 256 + threadIdx.x;       // dword id, 524288 total
  const int g    = idx & 7;
  const int lane = (idx >> 3) & 31;
  const int kt   = (idx >> 8) & (KTILES - 1);
  const int nt   = idx >> 13;
  const int n = lane & 15, half = lane >> 4;
  const int h = nt * 16 + n;
  const int k = kt * 32 + half * 16 + g * 2;
  const float* wr = W + (size_t)h * (2 * H_DIM) + H_DIM + k;
  _Float16* o = Bpack + (size_t)idx * 2;
  o[0] = (_Float16)wr[0];
  o[1] = (_Float16)wr[1];
}

// ---------------------------------------------------------------------------
// Kernel 2: fused GEMM + tanh + v-dot -> scores[B][S]   (deterministic)
// LDS A layout: [kt][row][32 halfwords] (k-linear inside the 64B row block).
// Lane fragment: bytes [kt*8192 + row*64 + half*32, +32).
// ---------------------------------------------------------------------------
__global__ __launch_bounds__(256) void attn_gemm_kernel(
    const float* __restrict__ enc, const _Float16* __restrict__ Bpack,
    const float* __restrict__ hb, const float* __restrict__ vvec,
    float* __restrict__ scores) {
  extern __shared__ unsigned char smem[];
  _Float16* Ah    = (_Float16*)smem;                      // 32*4096 halfwords = 256KB
  float*    wpart = (float*)(smem + MBLK * H_DIM * 2);    // [8 waves][128 rows]

  const int tid  = threadIdx.x;
  const int wave = tid >> 5, lane = tid & 31;
  const int n = lane & 15, half = lane >> 4;
  const int row0 = blockIdx.x * MBLK;

  // ---- Phase 1: stage enc rows into LDS, converting f32 -> f16 ----
  for (int c = tid; c < MBLK * 256; c += 256) {           // 256 float4 per row
    const int row = c >> 8, col4 = (c & 255) * 4;         // col4 = k, multiple of 4
    const float4 f = *(const float4*)(enc + (size_t)(row0 + row) * H_DIM + col4);
    union { _Float16 h[4]; uint2 u; } pk;
    pk.h[0] = (_Float16)f.x; pk.h[1] = (_Float16)f.y;
    pk.h[2] = (_Float16)f.z; pk.h[3] = (_Float16)f.w;
    const int kt = col4 >> 5, kl = col4 & 31;
    *(uint2*)((char*)Ah + (size_t)kt * 8192 + row * 64 + kl * 2) = pk.u;
  }
  __syncthreads();

  // ---- Phase 2: 8 M-tiles x 8 N-tiles per wave, K=1024 ----
  float partial[8][8];
#pragma unroll
  for (int mt = 0; mt < 8; ++mt)
#pragma unroll
    for (int r = 0; r < 8; ++r) partial[mt][r] = 0.0f;

  const char* AlaneBase = (const char*)Ah + n * 64 + half * 32;
  const int ntBeg = wave * 8;
  for (int ntd = 0; ntd < 8; ++ntd) {
    const int nt = ntBeg + ntd;
    const int hcol = nt * 16 + n;

    v8f acc[8];
#pragma unroll
    for (int mt = 0; mt < 8; ++mt) acc[mt] = (v8f){0, 0, 0, 0, 0, 0, 0, 0};

    const uint4* bp = (const uint4*)Bpack + (((size_t)nt * KTILES * 32 + lane) << 1);
    const char* abase = AlaneBase;
    for (int kt = 0; kt < KTILES; ++kt) {
      Frag16 bf;
      bf.q[0] = bp[0];
      bf.q[1] = bp[1];
      __builtin_prefetch((const void*)(bp + 64), 0, 1);   // next kt's fragment
      bp += 64;                                           // 32 lanes * 2 uint4
#pragma unroll
      for (int mt = 0; mt < 8; ++mt) {
        Frag16 af;                                        // DS immediate offsets
        af.q[0] = *(const uint4*)(abase + mt * 1024);
        af.q[1] = *(const uint4*)(abase + mt * 1024 + 16);
        acc[mt] = __builtin_amdgcn_wmma_f32_16x16x32_f16(
            false, af.h, false, bf.h, (short)0, acc[mt], false, false);
      }
      abase += 8192;                                      // next kt block
    }

    // epilogue: tanh + v-weighted accumulation (D row M = r + 8*half, col = n)
    const float vh = vvec[hcol];
#pragma unroll
    for (int mt = 0; mt < 8; ++mt) {
#pragma unroll
      for (int r = 0; r < 8; ++r) {
        const int b = r + 8 * half;                       // 16-row tile == one s
        const float e = tanhf(acc[mt][r] + hb[b * H_DIM + hcol]);
        partial[mt][r] += e * vh;
      }
    }
  }

  // ---- reduce across the 16 lanes sharing a row, then across waves ----
#pragma unroll
  for (int mt = 0; mt < 8; ++mt) {
#pragma unroll
    for (int r = 0; r < 8; ++r) {
      float val = partial[mt][r];
      val += __shfl_xor(val, 1, 32);
      val += __shfl_xor(val, 2, 32);
      val += __shfl_xor(val, 4, 32);
      val += __shfl_xor(val, 8, 32);
      if (n == 0) wpart[wave * MBLK + mt * 16 + r + 8 * half] = val;
    }
  }
  __syncthreads();

  if (tid < MBLK) {
    float sum = 0.0f;
#pragma unroll
    for (int w = 0; w < 8; ++w) sum += wpart[w * MBLK + tid];
    const int gr = row0 + tid;                            // gr = s*16 + b
    scores[(gr & 15) * S_DIM + (gr >> 4)] = sum;          // scores[b][s]
  }
}

// ---------------------------------------------------------------------------
// Kernel 3: softmax over S per batch row -> attn[B][S]
// ---------------------------------------------------------------------------
__global__ __launch_bounds__(256) void softmax_kernel(
    const float* __restrict__ scores, float* __restrict__ attn) {
  __shared__ float red[256];
  const int b = blockIdx.x, tid = threadIdx.x;
  const float* row = scores + (size_t)b * S_DIM;

  float x[8];
  float m = -INFINITY;
#pragma unroll
  for (int i = 0; i < 8; ++i) { x[i] = row[tid + i * 256]; m = fmaxf(m, x[i]); }
  red[tid] = m; __syncthreads();
  for (int off = 128; off > 0; off >>= 1) {
    if (tid < off) red[tid] = fmaxf(red[tid], red[tid + off]);
    __syncthreads();
  }
  m = red[0]; __syncthreads();

  float sum = 0.0f;
#pragma unroll
  for (int i = 0; i < 8; ++i) { x[i] = expf(x[i] - m); sum += x[i]; }
  red[tid] = sum; __syncthreads();
  for (int off = 128; off > 0; off >>= 1) {
    if (tid < off) red[tid] += red[tid + off];
    __syncthreads();
  }
  const float inv = 1.0f / red[0];
#pragma unroll
  for (int i = 0; i < 8; ++i) attn[(size_t)b * S_DIM + tid + i * 256] = x[i] * inv;
}

// ---------------------------------------------------------------------------
// Kernel 4: ctx[b,h] = sum_s attn[b,s] * enc[s,b,h]   (coalesced over h)
// ---------------------------------------------------------------------------
__global__ __launch_bounds__(256) void context_kernel(
    const float* __restrict__ enc, const float* __restrict__ attn,
    float* __restrict__ ctx) {
  const int b = blockIdx.y;
  const int h = blockIdx.x * 256 + threadIdx.x;
  const float* e = enc + (size_t)b * H_DIM + h;
  const float* a = attn + (size_t)b * S_DIM;
  float acc = 0.0f;
#pragma unroll 4
  for (int s = 0; s < S_DIM; ++s)
    acc += a[s] * e[(size_t)s * (B_DIM * H_DIM)];
  ctx[(size_t)b * H_DIM + h] = acc;
}

// ---------------------------------------------------------------------------
extern "C" void kernel_launch(void* const* d_in, const int* in_sizes, int n_in,
                              void* d_out, int out_size, void* d_ws, size_t ws_size,
                              hipStream_t stream) {
  const float* hidden = (const float*)d_in[0];   // [1,B,H]
  const float* enc    = (const float*)d_in[1];   // [S,B,H]
  const float* W      = (const float*)d_in[2];   // [H,2H]
  const float* bias   = (const float*)d_in[3];   // [H]
  const float* v      = (const float*)d_in[4];   // [H]

  float* out  = (float*)d_out;
  float* ctx  = out;                     // [B,H]  = 16384 floats
  float* attn = out + B_DIM * H_DIM;     // [B,S]  = 32768 floats

  char* ws = (char*)d_ws;
  float*    hb     = (float*)ws;                              // 64 KB
  _Float16* Bpack  = (_Float16*)(ws + 65536);                 // 2 MB
  float*    scores = (float*)(ws + 65536 + 2097152);          // 128 KB

  hbias_kernel<<<64, 256, 0, stream>>>(hidden, W, bias, hb);
  packB_kernel<<<2048, 256, 0, stream>>>(W, Bpack);

  const size_t lds_bytes = (size_t)MBLK * H_DIM * 2 + 8 * MBLK * 4;  // 266240
  attn_gemm_kernel<<<256, 256, lds_bytes, stream>>>(enc, Bpack, hb, v, scores);

  softmax_kernel<<<16, 256, 0, stream>>>(scores, attn);

  dim3 cgrid(H_DIM / 256, B_DIM);
  context_kernel<<<cgrid, 256, 0, stream>>>(enc, attn, ctx);
}
